// Mamba2LMHeadModel_50148038148752
// MI455X (gfx1250) — compile-verified, matrix operand order checked by reading, not run on hardware
//
#include <hip/hip_runtime.h>
#include <hip/hip_bf16.h>

// ---- model constants ----
#define D_MODEL   1024
#define N_LAYER   4
#define VOCAB     50288
#define D_STATE   128
#define D_CONV    4
#define HEADDIM   64
#define CHUNK     64
#define D_INNER   2048
#define NHEADS    32
#define D_IN_PROJ 4384       // 2*D_INNER + 2*D_STATE + NHEADS
#define CONV_DIM  2304       // D_INNER + 2*D_STATE
#define SEQLEN    1024
#define NCHUNKS   16

typedef __attribute__((ext_vector_type(16))) _Float16 v16h;
typedef __attribute__((ext_vector_type(8)))  float    v8f;

// =====================================================================
// WMMA GEMM:  D[M,N] = A[M,K] @ B[N,K]^T (+ optional residual Cres[M,N])
// One wave32 computes a 32(M) x 64(N) tile: 2 A fragments x 4 B fragments
// -> 8 v_wmma_f32_16x16x32_f16 per 32-wide K step. All 24 b128 raw loads
// of a K step are issued before any conversion so they overlap (no
// per-load s_wait_loadcnt 0x0). fp32 -> f16 conversion in registers,
// fp32 accumulation.
// Requires M%32==0, N%16==0, K%32==0 (true for all call sites here).
// =====================================================================
struct Raw16 { float4 a, b, c, d; };

__device__ __forceinline__ Raw16 load_raw(const float* __restrict__ row, int kb)
{
    // lane owns K = {kb..kb+7, kb+16..kb+23}; 32-byte aligned here.
    Raw16 r;
    r.a = *(const float4*)(row + kb);
    r.b = *(const float4*)(row + kb + 4);
    r.c = *(const float4*)(row + kb + 16);
    r.d = *(const float4*)(row + kb + 20);
    return r;
}

__device__ __forceinline__ v16h cvt_frag(const Raw16& t)
{
    v16h f;
    f[0]  = (_Float16)t.a.x; f[1]  = (_Float16)t.a.y;
    f[2]  = (_Float16)t.a.z; f[3]  = (_Float16)t.a.w;
    f[4]  = (_Float16)t.b.x; f[5]  = (_Float16)t.b.y;
    f[6]  = (_Float16)t.b.z; f[7]  = (_Float16)t.b.w;
    f[8]  = (_Float16)t.c.x; f[9]  = (_Float16)t.c.y;
    f[10] = (_Float16)t.c.z; f[11] = (_Float16)t.c.w;
    f[12] = (_Float16)t.d.x; f[13] = (_Float16)t.d.y;
    f[14] = (_Float16)t.d.z; f[15] = (_Float16)t.d.w;
    return f;
}

#define WMMA_F16(a, b, c) \
    __builtin_amdgcn_wmma_f32_16x16x32_f16(false, (a), false, (b), (short)0, (c), false, false)

template <bool HAS_RES>
__global__ void gemm_wmma_kernel(const float* __restrict__ A,
                                 const float* __restrict__ B,
                                 const float* __restrict__ Cres,
                                 float* __restrict__ D,
                                 int M, int N, int K)
{
    int gwave  = (int)((blockIdx.x * blockDim.x + threadIdx.x) >> 5);
    int lane   = threadIdx.x & 31;
    int tilesN = N >> 4;                  // 16-wide N tiles
    int strpN  = (tilesN + 3) >> 2;       // 64-wide strips
    int tilesM = M >> 5;                  // 32-high M tiles
    if (gwave >= tilesM * strpN) return;
    int tm = gwave / strpN;
    int ts = gwave - tm * strpN;
    int m0 = tm << 5;
    int n0 = ts << 6;                     // strip base (4 x 16)

    int r  = lane & 15;   // row of A / col of B owned by this lane
    int hi = lane >> 4;   // K-half selector per ISA fragment layout

    const float* arow0 = A + (size_t)(m0 + r) * K;
    const float* arow1 = A + (size_t)(m0 + 16 + r) * K;

    bool         valid[4];
    const float* brow[4];
#pragma unroll
    for (int j = 0; j < 4; ++j) {
        int nj   = n0 + (j << 4);
        valid[j] = (nj < N);
        brow[j]  = B + (size_t)((valid[j] ? nj : 0) + r) * K;  // clamp OOB strip
    }

    v8f acc00 = {}, acc01 = {}, acc02 = {}, acc03 = {};
    v8f acc10 = {}, acc11 = {}, acc12 = {}, acc13 = {};

    for (int k = 0; k < K; k += 32) {
        int kb = k + (hi << 3);
        if (k + 64 < K) {
            __builtin_prefetch(arow0 + kb + 64, 0, 0);
            __builtin_prefetch(arow1 + kb + 64, 0, 0);
        }
        // ---- issue ALL raw loads for this K step first ----
        Raw16 rA0 = load_raw(arow0,  kb);
        Raw16 rA1 = load_raw(arow1,  kb);
        Raw16 rB0 = load_raw(brow[0], kb);
        Raw16 rB1 = load_raw(brow[1], kb);
        Raw16 rB2 = load_raw(brow[2], kb);
        Raw16 rB3 = load_raw(brow[3], kb);
        // ---- convert + matrix ops ----
        v16h a0 = cvt_frag(rA0);
        v16h a1 = cvt_frag(rA1);
        v16h b;
        b = cvt_frag(rB0); acc00 = WMMA_F16(a0, b, acc00); acc10 = WMMA_F16(a1, b, acc10);
        b = cvt_frag(rB1); acc01 = WMMA_F16(a0, b, acc01); acc11 = WMMA_F16(a1, b, acc11);
        b = cvt_frag(rB2); acc02 = WMMA_F16(a0, b, acc02); acc12 = WMMA_F16(a1, b, acc12);
        b = cvt_frag(rB3); acc03 = WMMA_F16(a0, b, acc03); acc13 = WMMA_F16(a1, b, acc13);
    }

    // D tile store: VGPR i holds M = mbase + i + 8*hi, N = n_tile + (lane&15)
    v8f accs[2][4] = {{acc00, acc01, acc02, acc03}, {acc10, acc11, acc12, acc13}};
#pragma unroll
    for (int jm = 0; jm < 2; ++jm) {
#pragma unroll
        for (int j = 0; j < 4; ++j) {
            if (!valid[j]) continue;
            int n = n0 + (j << 4) + r;
#pragma unroll
            for (int i = 0; i < 8; ++i) {
                int m = m0 + (jm << 4) + i + (hi << 3);
                size_t o = (size_t)m * N + n;
                float v = accs[jm][j][i];
                if (HAS_RES) v += Cres[o];
                D[o] = v;
            }
        }
    }
}

static void launch_gemm(const float* A, const float* B, const float* Cres,
                        float* D, int M, int N, int K, hipStream_t stream)
{
    int strpN  = ((N >> 4) + 3) >> 2;
    int waves  = (M >> 5) * strpN;
    int blocks = (waves + 7) / 8;          // 8 waves (256 threads) per block
    if (Cres)
        gemm_wmma_kernel<true><<<blocks, 256, 0, stream>>>(A, B, Cres, D, M, N, K);
    else
        gemm_wmma_kernel<false><<<blocks, 256, 0, stream>>>(A, B, Cres, D, M, N, K);
}

// =====================================================================
// Elementwise / reduction kernels
// =====================================================================
__device__ __forceinline__ float silu_f(float x) {
    return x / (1.0f + __expf(-x));
}

__global__ void embed_kernel(const int* __restrict__ ids,
                             const float* __restrict__ emb,
                             float* __restrict__ h)
{
    int t = blockIdx.x;
    const float* src = emb + (size_t)ids[t] * D_MODEL;
    for (int c = threadIdx.x; c < D_MODEL; c += blockDim.x)
        h[(size_t)t * D_MODEL + c] = src[c];
}

// rmsnorm with optional silu(z) gate. strides in elements.
__global__ void rmsnorm_kernel(const float* __restrict__ x, int xstride,
                               const float* __restrict__ w,
                               const float* __restrict__ z, int zstride,
                               float* __restrict__ out, int ostride, int ncols)
{
    __shared__ float sbuf[256];
    int row = blockIdx.x;
    const float* xr = x + (size_t)row * xstride;
    const float* zr = z ? (z + (size_t)row * zstride) : nullptr;

    float lsum = 0.f;
    for (int c = threadIdx.x; c < ncols; c += blockDim.x) {
        float v = xr[c];
        if (zr) v *= silu_f(zr[c]);
        lsum += v * v;
    }
    sbuf[threadIdx.x] = lsum;
    __syncthreads();
    for (int off = 128; off > 0; off >>= 1) {
        if ((int)threadIdx.x < off) sbuf[threadIdx.x] += sbuf[threadIdx.x + off];
        __syncthreads();
    }
    float scale = rsqrtf(sbuf[0] / (float)ncols + 1e-5f);
    for (int c = threadIdx.x; c < ncols; c += blockDim.x) {
        float v = xr[c];
        if (zr) v *= silu_f(zr[c]);
        out[(size_t)row * ostride + c] = v * scale * w[c];
    }
}

__global__ void dt_kernel(const float* __restrict__ zxbcdt,
                          const float* __restrict__ dt_bias,
                          float* __restrict__ dt)
{
    int i = blockIdx.x * blockDim.x + threadIdx.x;   // t*NHEADS + h
    if (i >= SEQLEN * NHEADS) return;
    int t = i >> 5, h = i & 31;
    float v = zxbcdt[(size_t)t * D_IN_PROJ + (D_IN_PROJ - NHEADS) + h] + dt_bias[h];
    dt[i] = (v > 20.f) ? v : log1pf(__expf(v));
}

// causal depthwise conv (kernel 4) + bias + silu over xBC slice of zxbcdt
__global__ void conv_kernel(const float* __restrict__ zxbcdt,
                            const float* __restrict__ w,
                            const float* __restrict__ b,
                            float* __restrict__ xconv)
{
    int i = blockIdx.x * blockDim.x + threadIdx.x;   // t*CONV_DIM + c
    if (i >= SEQLEN * CONV_DIM) return;
    int t = i / CONV_DIM, c = i - t * CONV_DIM;
    float acc = b[c];
#pragma unroll
    for (int j = 0; j < D_CONV; ++j) {
        int tt = t - (D_CONV - 1) + j;
        if (tt >= 0)
            acc += zxbcdt[(size_t)tt * D_IN_PROJ + D_INNER + c] * w[c * D_CONV + j];
    }
    xconv[i] = silu_f(acc);
}

// per (head, chunk) cumulative sum of A*dt over the chunk
__global__ void acs_kernel(const float* __restrict__ A_log,
                           const float* __restrict__ dt,
                           float* __restrict__ a_cs,     // [32][16][64]
                           float* __restrict__ a_last)   // [32][16]
{
    int i = blockIdx.x * blockDim.x + threadIdx.x;       // h*16 + c
    if (i >= NHEADS * NCHUNKS) return;
    int h = i >> 4, c = i & 15;
    float A = -__expf(A_log[h]);
    float s = 0.f;
    for (int l = 0; l < CHUNK; ++l) {
        int t = c * CHUNK + l;
        s += A * dt[t * NHEADS + h];
        a_cs[(size_t)i * CHUNK + l] = s;
    }
    a_last[i] = s;
}

// states[c,h,p,n] = sum_l B[t,n] * exp(acs[63]-acs[l]) * x[t,h,p]*dt[t,h]
__global__ void states_kernel(const float* __restrict__ xconv,
                              const float* __restrict__ dt,
                              const float* __restrict__ a_cs,
                              float* __restrict__ states)
{
    int c = blockIdx.x, h = blockIdx.y;
    const float* acs = a_cs + ((size_t)h * NCHUNKS + c) * CHUNK;
    float alast = acs[CHUNK - 1];
    for (int idx = threadIdx.x; idx < HEADDIM * D_STATE; idx += blockDim.x) {
        int p = idx >> 7, n = idx & 127;
        float acc = 0.f;
        for (int l = 0; l < CHUNK; ++l) {
            int t = c * CHUNK + l;
            float bv = xconv[(size_t)t * CONV_DIM + D_INNER + n];
            float xv = xconv[(size_t)t * CONV_DIM + h * HEADDIM + p];
            acc += bv * __expf(alast - acs[l]) * xv * dt[t * NHEADS + h];
        }
        states[((((size_t)c * NHEADS + h) * HEADDIM) + p) * D_STATE + n] = acc;
    }
}

// inter-chunk recurrence: sprev[c] = state entering chunk c
__global__ void scan_kernel(const float* __restrict__ states,
                            const float* __restrict__ a_last,
                            float* __restrict__ sprev)
{
    int h = blockIdx.x;
    for (int idx = threadIdx.x; idx < HEADDIM * D_STATE; idx += blockDim.x) {
        float carry = 0.f;
        for (int c = 0; c < NCHUNKS; ++c) {
            size_t o = (((size_t)c * NHEADS + h) * (HEADDIM * D_STATE)) + idx;
            sprev[o] = carry;
            carry = carry * __expf(a_last[h * NCHUNKS + c]) + states[o];
        }
    }
}

// CB[c,l,s] = dot(C[c,l,:], B[c,s,:]) over D_STATE
__global__ void cb_kernel(const float* __restrict__ xconv,
                          float* __restrict__ cb)
{
    int c = blockIdx.x;
    for (int idx = threadIdx.x; idx < CHUNK * CHUNK; idx += blockDim.x) {
        int l = idx >> 6, s = idx & 63;
        const float* Crow = xconv + (size_t)(c * CHUNK + l) * CONV_DIM + D_INNER + D_STATE;
        const float* Brow = xconv + (size_t)(c * CHUNK + s) * CONV_DIM + D_INNER;
        float acc = 0.f;
        for (int n = 0; n < D_STATE; ++n) acc += Crow[n] * Brow[n];
        cb[((size_t)c * CHUNK + l) * CHUNK + s] = acc;
    }
}

// Y = Y_diag + Y_off + x*D
__global__ void y_kernel(const float* __restrict__ xconv,
                         const float* __restrict__ dt,
                         const float* __restrict__ a_cs,
                         const float* __restrict__ cb,
                         const float* __restrict__ sprev,
                         const float* __restrict__ Dp,
                         float* __restrict__ y)
{
    int c = blockIdx.x, h = blockIdx.y;
    const float* acs = a_cs + ((size_t)h * NCHUNKS + c) * CHUNK;
    for (int idx = threadIdx.x; idx < CHUNK * HEADDIM; idx += blockDim.x) {
        int l = idx >> 6, p = idx & 63;
        int t = c * CHUNK + l;
        float acl = acs[l];
        float acc = 0.f;
        for (int s = 0; s <= l; ++s) {
            int ts = c * CHUNK + s;
            float xv = xconv[(size_t)ts * CONV_DIM + h * HEADDIM + p] * dt[ts * NHEADS + h];
            acc += cb[((size_t)c * CHUNK + l) * CHUNK + s] * __expf(acl - acs[s]) * xv;
        }
        const float* Crow = xconv + (size_t)t * CONV_DIM + D_INNER + D_STATE;
        const float* Sp   = sprev + (((size_t)c * NHEADS + h) * HEADDIM + p) * D_STATE;
        float off = 0.f;
        for (int n = 0; n < D_STATE; ++n) off += Crow[n] * Sp[n];
        acc += __expf(acl) * off;
        acc += xconv[(size_t)t * CONV_DIM + h * HEADDIM + p] * Dp[h];
        y[(size_t)t * D_INNER + h * HEADDIM + p] = acc;
    }
}

// =====================================================================
// Host orchestration
// =====================================================================
extern "C" void kernel_launch(void* const* d_in, const int* in_sizes, int n_in,
                              void* d_out, int out_size, void* d_ws, size_t ws_size,
                              hipStream_t stream)
{
    const int*   ids          = (const int*)  d_in[0];
    const float* emb          = (const float*)d_in[1];
    const float* norm_w       = (const float*)d_in[2];
    const float* in_proj_w    = (const float*)d_in[3];
    const float* conv_w       = (const float*)d_in[4];
    const float* conv_b       = (const float*)d_in[5];
    const float* dt_bias      = (const float*)d_in[6];
    const float* A_log        = (const float*)d_in[7];
    const float* Dp           = (const float*)d_in[8];
    const float* mixer_norm_w = (const float*)d_in[9];
    const float* out_proj_w   = (const float*)d_in[10];
    const float* norm_f_w     = (const float*)d_in[11];
    float*       out          = (float*)d_out;

    float* ws = (float*)d_ws;
    float* h       = ws; ws += (size_t)SEQLEN * D_MODEL;
    float* u       = ws; ws += (size_t)SEQLEN * D_MODEL;
    float* zxbcdt  = ws; ws += (size_t)SEQLEN * D_IN_PROJ;
    float* xconv   = ws; ws += (size_t)SEQLEN * CONV_DIM;
    float* dt      = ws; ws += (size_t)SEQLEN * NHEADS;
    float* a_cs    = ws; ws += (size_t)NHEADS * NCHUNKS * CHUNK;
    float* a_last  = ws; ws += (size_t)NHEADS * NCHUNKS;
    float* states  = ws; ws += (size_t)NCHUNKS * NHEADS * HEADDIM * D_STATE;
    float* sprev   = ws; ws += (size_t)NCHUNKS * NHEADS * HEADDIM * D_STATE;
    float* cb      = ws; ws += (size_t)NCHUNKS * CHUNK * CHUNK;
    float* y       = ws; ws += (size_t)SEQLEN * D_INNER;
    float* y2      = ws; ws += (size_t)SEQLEN * D_INNER;

    embed_kernel<<<SEQLEN, 256, 0, stream>>>(ids, emb, h);

    for (int l = 0; l < N_LAYER; ++l) {
        rmsnorm_kernel<<<SEQLEN, 256, 0, stream>>>(
            h, D_MODEL, norm_w + (size_t)l * D_MODEL, nullptr, 0, u, D_MODEL, D_MODEL);

        launch_gemm(u, in_proj_w + (size_t)l * D_IN_PROJ * D_MODEL, nullptr,
                    zxbcdt, SEQLEN, D_IN_PROJ, D_MODEL, stream);

        dt_kernel<<<(SEQLEN * NHEADS + 255) / 256, 256, 0, stream>>>(
            zxbcdt, dt_bias + (size_t)l * NHEADS, dt);

        conv_kernel<<<(SEQLEN * CONV_DIM + 255) / 256, 256, 0, stream>>>(
            zxbcdt, conv_w + (size_t)l * CONV_DIM * D_CONV,
            conv_b + (size_t)l * CONV_DIM, xconv);

        acs_kernel<<<(NHEADS * NCHUNKS + 255) / 256, 256, 0, stream>>>(
            A_log + (size_t)l * NHEADS, dt, a_cs, a_last);

        states_kernel<<<dim3(NCHUNKS, NHEADS), 256, 0, stream>>>(xconv, dt, a_cs, states);
        scan_kernel<<<NHEADS, 256, 0, stream>>>(states, a_last, sprev);
        cb_kernel<<<NCHUNKS, 256, 0, stream>>>(xconv, cb);
        y_kernel<<<dim3(NCHUNKS, NHEADS), 256, 0, stream>>>(
            xconv, dt, a_cs, cb, sprev, Dp + (size_t)l * NHEADS, y);

        rmsnorm_kernel<<<SEQLEN, 256, 0, stream>>>(
            y, D_INNER, mixer_norm_w + (size_t)l * D_INNER,
            zxbcdt /* z = cols [0,2048) */, D_IN_PROJ, y2, D_INNER, D_INNER);

        launch_gemm(y2, out_proj_w + (size_t)l * D_MODEL * D_INNER, h /* residual */,
                    h, SEQLEN, D_MODEL, D_INNER, stream);
    }

    rmsnorm_kernel<<<SEQLEN, 256, 0, stream>>>(
        h, D_MODEL, norm_f_w, nullptr, 0, u, D_MODEL, D_MODEL);

    launch_gemm(u, emb, nullptr, out, SEQLEN, VOCAB, D_MODEL, stream);
}
